// MyModel_48670569399069
// MI455X (gfx1250) — matrix operand order repaired and checked by reading, not compiled, not run on hardware
//
#include <hip/hip_runtime.h>
#include <hip/hip_bf16.h>

typedef __attribute__((ext_vector_type(16))) _Float16 v16h;
typedef __attribute__((ext_vector_type(8)))  float    v8f;

// Problem constants (from reference):
//   N_NODE=200000, DIM=64, N_FACTOR=4, N_LAYER=2, K=32, B=1024
// node row = N_FACTOR*DIM = 256 floats (1 KB)
// One block  = one (tower, batch, layer): M=128 rows (m = f*32 + k), K=128, N=64.
// Grid = 2 * 1024 * 2 = 4096 blocks, 256 threads (8 wave32 waves).

__global__ void __launch_bounds__(256)
kgat_tower_kernel(const float* __restrict__ node_emb,
                  const float* __restrict__ W1,   // (128, 64) row-major
                  const float* __restrict__ b1,   // (64,)
                  const float* __restrict__ W2,   // (64, 1)
                  const int*   __restrict__ users,
                  const int*   __restrict__ items,
                  const int*   __restrict__ users_triple,  // (B,3,2,32) int32
                  const int*   __restrict__ items_triple,
                  float*       __restrict__ out)           // (2,3,B,4,64)
{
    __shared__ _Float16 hA[128][64];   // x[:, 0:64]  = h rows  (f16)
    __shared__ _Float16 tA[128][64];   // x[:, 64:128]= t rows  (f16)
    __shared__ _Float16 w1s[64][128];  // W1 transposed: w1s[n][k] = W1[k][n]
    __shared__ float    b1s[64];
    __shared__ float    w2s[64];
    __shared__ float    lg[128];       // logits per row m = f*32+k
    __shared__ float    wg[128];       // softmax weights
    __shared__ int      hidx[32];
    __shared__ int      tidx[32];

    const int tid = threadIdx.x;
    const int gid = blockIdx.x;
    const int tw  = gid >> 11;          // 0 = users tower, 1 = items tower
    const int rem = gid & 2047;
    const int b   = rem >> 1;           // batch index
    const int l   = rem & 1;            // layer index

    const int* triple = tw ? items_triple : users_triple;
    const int* center = tw ? items       : users;

    // ---- stage indices / bias / W2 ----
    if (tid < 32) {
        hidx[tid] = triple[b * 192 + 0 * 64 + l * 32 + tid];  // head indices
        tidx[tid] = triple[b * 192 + 2 * 64 + l * 32 + tid];  // tail indices
    } else if (tid < 96) {
        const int n = tid - 32;
        b1s[n] = b1[n];
        w2s[n] = W2[n];
    }
    __syncthreads();

    // ---- gather h/t rows (float4, 1KB per node row) -> f16 LDS; stage W1^T ----
    for (int i = tid; i < 2048; i += 256) {
        const int m = i >> 4;           // row 0..127   (m = f*32 + k)
        const int q = (i & 15) * 4;     // float column within row
        const int k = m & 31;
        const int f = m >> 5;
        const float4 hv = *reinterpret_cast<const float4*>(
            node_emb + (size_t)hidx[k] * 256 + f * 64 + q);
        const float4 tv = *reinterpret_cast<const float4*>(
            node_emb + (size_t)tidx[k] * 256 + f * 64 + q);
        hA[m][q + 0] = (_Float16)hv.x; hA[m][q + 1] = (_Float16)hv.y;
        hA[m][q + 2] = (_Float16)hv.z; hA[m][q + 3] = (_Float16)hv.w;
        tA[m][q + 0] = (_Float16)tv.x; tA[m][q + 1] = (_Float16)tv.y;
        tA[m][q + 2] = (_Float16)tv.z; tA[m][q + 3] = (_Float16)tv.w;
        // W1 row m (= K index), cols q..q+3 (= N index) -> transposed store
        const float4 wv = *reinterpret_cast<const float4*>(W1 + m * 64 + q);
        w1s[q + 0][m] = (_Float16)wv.x; w1s[q + 1][m] = (_Float16)wv.y;
        w1s[q + 2][m] = (_Float16)wv.z; w1s[q + 3][m] = (_Float16)wv.w;
    }
    __syncthreads();

    // ---- GEMM: hid = relu([h;t] @ W1 + b1), fused logits = hid . W2 ----
    const int lane = tid & 31;
    const int w    = tid >> 5;          // wave id 0..7 -> rows [16w, 16w+16)
    const int nl   = lane & 15;
    const int hi   = lane >> 4;
    const int m    = 16 * w + nl;       // A-operand row for this lane

    v8f c[4];
    #pragma unroll
    for (int nt = 0; nt < 4; ++nt) c[nt] = (v8f){0.f,0.f,0.f,0.f,0.f,0.f,0.f,0.f};

    #pragma unroll
    for (int s = 0; s < 4; ++s) {                       // K-step: 32 per step
        const _Float16 (*src)[64] = (s < 2) ? hA : tA;  // K 0-63 from h, 64-127 from t
        const int kbase = (s & 1) * 32;

        // A operand: ISA 16-bit 16x32 layout
        // lanes 0-15: K in {0..7, 16..23}; lanes 16-31: K in {8..15, 24..31}
        v16h a;
        #pragma unroll
        for (int e = 0; e < 16; ++e) {
            const int p    = e >> 1;
            const int koff = ((p < 4) ? (2 * p) : (16 + 2 * (p - 4))) + 8 * hi + (e & 1);
            a[e] = src[m][kbase + koff];
        }

        #pragma unroll
        for (int nt = 0; nt < 4; ++nt) {
            // B operand: lanes 0-15: K in {0..7, 16..23}; lanes 16-31: K in {8..15, 24..31}
            v16h bm;
            #pragma unroll
            for (int e = 0; e < 16; ++e) {
                const int p  = e >> 1;
                const int kr = 32 * s + ((p < 4) ? (2 * p) : (16 + 2 * (p - 4))) + 8 * hi + (e & 1);
                bm[e] = w1s[nt * 16 + nl][kr];
            }
            c[nt] = __builtin_amdgcn_wmma_f32_16x16x32_f16(
                        false, a, false, bm, (short)0, c[nt], false, false);
        }
    }

    // ---- epilogue: +b1, relu, dot with W2; reduce across N (16 lanes) ----
    // D layout: VGPR r <-> M = 16w + r + 8*hi, N = nt*16 + nl
    float part[8] = {0.f,0.f,0.f,0.f,0.f,0.f,0.f,0.f};
    #pragma unroll
    for (int nt = 0; nt < 4; ++nt) {
        const int   N  = nt * 16 + nl;
        const float bb = b1s[N];
        const float ww = w2s[N];
        #pragma unroll
        for (int r = 0; r < 8; ++r) {
            float v = c[nt][r] + bb;
            v = v > 0.f ? v : 0.f;
            part[r] += v * ww;
        }
    }
    #pragma unroll
    for (int off = 1; off < 16; off <<= 1) {
        #pragma unroll
        for (int r = 0; r < 8; ++r)
            part[r] += __shfl_xor(part[r], off, 32);
    }
    if (nl == 0) {
        #pragma unroll
        for (int r = 0; r < 8; ++r)
            lg[16 * w + 8 * hi + r] = part[r];
    }
    __syncthreads();

    // ---- softmax over k (32) per factor f (b2 dropped: shift-invariant) ----
    if (tid < 4) {
        const int base = tid * 32;
        float mx = lg[base];
        for (int k = 1; k < 32; ++k) mx = fmaxf(mx, lg[base + k]);
        float sum = 0.f;
        for (int k = 0; k < 32; ++k) {
            const float e = __expf(lg[base + k] - mx);
            wg[base + k] = e;
            sum += e;
        }
        const float inv = 1.f / sum;
        for (int k = 0; k < 32; ++k) wg[base + k] *= inv;
    }
    __syncthreads();

    // ---- weighted sum of t rows -> out[tw, 1+l, b, f, :] ----
    {
        const int f = tid >> 6;
        const int d = tid & 63;
        float acc = 0.f;
        #pragma unroll
        for (int k = 0; k < 32; ++k)
            acc += wg[f * 32 + k] * (float)tA[f * 32 + k][d];
        out[(((size_t)(tw * 3 + 1 + l)) * 1024 + b) * 256 + tid] = acc;
    }

    // ---- origin row -> out[tw, 0, b, :, :] (write once: layer-0 blocks) ----
    if (l == 0) {
        const int ci = center[b];
        out[(((size_t)(tw * 3)) * 1024 + b) * 256 + tid] =
            node_emb[(size_t)ci * 256 + tid];
    }
}

extern "C" void kernel_launch(void* const* d_in, const int* in_sizes, int n_in,
                              void* d_out, int out_size, void* d_ws, size_t ws_size,
                              hipStream_t stream) {
    (void)in_sizes; (void)n_in; (void)out_size; (void)d_ws; (void)ws_size;
    const float* node_emb = (const float*)d_in[0];
    // d_in[1] = relation_emb : unused by the reference computation
    const float* W1 = (const float*)d_in[2];
    const float* b1 = (const float*)d_in[3];
    const float* W2 = (const float*)d_in[4];
    // d_in[5] = b2 : uniform logit shift, softmax-invariant -> dropped exactly
    const int* users        = (const int*)d_in[6];
    const int* items        = (const int*)d_in[7];
    const int* users_triple = (const int*)d_in[8];
    const int* items_triple = (const int*)d_in[9];
    float* out = (float*)d_out;

    dim3 grid(4096), block(256);
    hipLaunchKernelGGL(kgat_tower_kernel, grid, block, 0, stream,
                       node_emb, W1, b1, W2, users, items,
                       users_triple, items_triple, out);
}
